// MeSH_GCN_45200235823785
// MI455X (gfx1250) — matrix operand order repaired and checked by reading, not compiled, not run on hardware
//
#include <hip/hip_runtime.h>
#include <hip/hip_bf16.h>
#include <math.h>

// ---------------- constants ----------------
#define CB   4        // batch
#define CL   512      // seq len
#define CD   200      // feature dim
#define CDP  224      // padded D (mult of 32)
#define CN   28000    // nodes (28000/16 = 1750 exact)
#define CNK  128      // conv kernels
#define CHG  512      // GCN hidden
#define C3NK 384      // 3*NK
#define C2D  400      // 2*D
#define C2DP 208      // padded 200 rows for Wg2t
#define KMAX 512      // max GEMM K depth (all calls: 224/384/512)

typedef __attribute__((ext_vector_type(16))) __bf16 v16bf;
typedef __attribute__((ext_vector_type(8)))  __bf16 v8bf;
typedef __attribute__((ext_vector_type(8)))  float  v8f;
typedef __attribute__((ext_vector_type(4)))  unsigned u32x4;
typedef __attribute__((ext_vector_type(8)))  int   i32x8;
typedef __attribute__((ext_vector_type(4)))  int   i32x4;

#if defined(__has_builtin)
#if __has_builtin(__builtin_amdgcn_tensor_load_to_lds)
#define TDM_OK 1
#else
#define TDM_OK 0
#endif
#else
#define TDM_OK 0
#endif

// ---------------- epilogue modes ----------------
enum { EP_F32 = 0, EP_BIAS_RELU_F32 = 1, EP_BIAS_GUARD_F32 = 2, EP_BF16T = 3, EP_XF = 4 };

struct EpArgs {
  float*        dstF;    int ldd;
  const float*  bias;
  int           guard;
  __bf16*       dstT;    int ldt; int colbase;
  const float*  h2;      const float* feat;
  float*        outacc;
};

#if TDM_OK
// TDM 2D tile load: copy tile [tile_rows x tile_k halves] (global row stride = row_stride
// halves, 2-byte elements) from global memory into contiguous LDS (LDS row stride = tile_k)
// at lds_off.  D# per ISA 8.3/8.4:
// group0 = {count=1, lds_addr, global_addr lo, global_addr hi | type=2}
// group1 = {data_size=2B, tensor_dim0/1 (remaining extents -> OOB reads return 0),
//           tile_dim0/1, tensor_dim0_stride}
__device__ __forceinline__ void tdm_load_2d(unsigned lds_off, const void* gptr,
                                            unsigned tile_k, unsigned tile_rows,
                                            unsigned rem_k, unsigned rem_rows,
                                            unsigned row_stride) {
  unsigned long long ga = (unsigned long long)(size_t)gptr;
  u32x4 g0;
  g0[0] = 1u;                                   // count=1 valid user descriptor
  g0[1] = lds_off;                              // LDS byte address
  g0[2] = (unsigned)ga;                         // global_addr[31:0]
  g0[3] = (unsigned)((ga >> 32) & 0x1FFFFFFull) | (2u << 30);  // addr[56:32] | type=2
  i32x8 g1;
  g1[0] = (int)(1u << 16);                                        // data_size=1 -> 2B
  g1[1] = (int)((rem_k & 0xFFFFu) << 16);                         // tensor_dim0[15:0]
  g1[2] = (int)((rem_k >> 16) | ((rem_rows & 0xFFFFu) << 16));    // dim0[31:16]|dim1[15:0]
  g1[3] = (int)((rem_rows >> 16) | ((tile_k & 0xFFFFu) << 16));   // dim1[31:16]|tile_dim0
  g1[4] = (int)(tile_rows & 0xFFFFu);                             // tile_dim1 (tile_dim2=0)
  g1[5] = (int)row_stride;                                        // tensor_dim0_stride lo
  g1[6] = 0;
  g1[7] = 0;
  i32x4 g2 = {0, 0, 0, 0};
  i32x4 g3 = {0, 0, 0, 0};
#if defined(__clang_major__) && (__clang_major__ >= 23)
  i32x8 g4 = {0, 0, 0, 0, 0, 0, 0, 0};
  __builtin_amdgcn_tensor_load_to_lds(g0, g1, g2, g3, g4, 0);
#else
  __builtin_amdgcn_tensor_load_to_lds(g0, g1, g2, g3, 0);
#endif
}
#endif

// Generic WMMA GEMM.  Block = 4 waves = one 16M x 64N macro-tile.
// A: M x K row-major bf16 (lda, K mult of 32 and <= KMAX, M mult of 16).
// Bt: N-major "B transposed" bf16: Bt[n*ldb + k].
// TDM path: wave0 stages the WHOLE A panel (16 x K) and B panel (64 x K) into LDS with
// two tensor_load_to_lds ops, one s_wait_tensorcnt + one barrier, then the K-loop is
// pure ds_load + wmma with no synchronization.  Ragged N handled by TDM OOB zero-fill
// (tensor_dim1 = remaining rows) + store guard.
template<int MODE>
__global__ __launch_bounds__(128)
void wmma_gemm_bf16(const __bf16* __restrict__ A, int lda,
                    const __bf16* __restrict__ Bt, int ldb,
                    int K, int Ncols, EpArgs ep) {
  const int lane = threadIdx.x & 31;
  const int wave = threadIdx.x >> 5;
  const int r    = lane & 15;
  const int hi   = lane >> 4;
  const int M0   = blockIdx.y * 16;
  const int NB   = blockIdx.x * 64;
  const int N0   = NB + wave * 16;
  v8f acc = {};

#if TDM_OK
  __shared__ __align__(16) __bf16 sA[16 * KMAX];   // 16 KB
  __shared__ __align__(16) __bf16 sB[64 * KMAX];   // 64 KB
  if (wave == 0) {
    const unsigned remRows = (unsigned)(Ncols > NB ? Ncols - NB : 0);
    tdm_load_2d((unsigned)(size_t)(void*)&sA[0], A + (size_t)M0 * lda,
                (unsigned)K, 16u, (unsigned)K, 16u, (unsigned)lda);
    tdm_load_2d((unsigned)(size_t)(void*)&sB[0], Bt + (size_t)NB * ldb,
                (unsigned)K, 64u, (unsigned)K, remRows, (unsigned)ldb);
    __builtin_amdgcn_s_wait_tensorcnt(0);
  }
  __syncthreads();
  // A frag (16x32, contiguous k order, same as B): lane r = row; lanes<16: K 0..15; lanes>=16: K 16..31
  const __bf16* sa = sA + r * K + hi * 16;
  // B frag (32x16): lane r = column; lanes<16: K 0..15; lanes>=16: K 16..31
  const __bf16* sb = sB + (wave * 16 + r) * K + hi * 16;
  for (int kk = 0; kk < K; kk += 32) {
    v8bf a0 = *(const v8bf*)(sa + kk);
    v8bf a1 = *(const v8bf*)(sa + kk + 8);
    v16bf a;
#pragma unroll
    for (int i = 0; i < 8; ++i) { a[i] = a0[i]; a[i + 8] = a1[i]; }
    v16bf b = *(const v16bf*)(sb + kk);
    acc = __builtin_amdgcn_wmma_f32_16x16x32_bf16(false, a, false, b,
                                                  (short)0, acc, false, false);
  }
#else
  int brow_i = N0 + r; if (brow_i >= Ncols) brow_i = Ncols - 1;   // clamp; result discarded
  const __bf16* arow = A  + (size_t)(M0 + r) * lda + hi * 16;
  const __bf16* brow = Bt + (size_t)brow_i * ldb + hi * 16;
  for (int kk = 0; kk < K; kk += 32) {
    v8bf a0 = *(const v8bf*)(arow + kk);
    v8bf a1 = *(const v8bf*)(arow + kk + 8);
    v16bf a;
#pragma unroll
    for (int i = 0; i < 8; ++i) { a[i] = a0[i]; a[i + 8] = a1[i]; }
    v16bf b = *(const v16bf*)(brow + kk);
    acc = __builtin_amdgcn_wmma_f32_16x16x32_bf16(false, a, false, b,
                                                  (short)0, acc, false, false);
  }
#endif

  const int col = N0 + r;
  if (col < Ncols) {
#pragma unroll
    for (int i = 0; i < 8; ++i) {
      const int row = M0 + i + hi * 8;   // C/D: VGPR i -> M=i (+8 hi half), N=lane%16
      float v = acc[i];
      if (MODE == EP_F32) {
        ep.dstF[(size_t)row * ep.ldd + col] = v;
      } else if (MODE == EP_BIAS_RELU_F32) {
        v += ep.bias[col]; v = v > 0.f ? v : 0.f;
        ep.dstF[(size_t)row * ep.ldd + col] = v;
      } else if (MODE == EP_BIAS_GUARD_F32) {
        if (col < ep.guard)
          ep.dstF[(size_t)row * ep.ldd + col] = v + ep.bias[col];
      } else if (MODE == EP_BF16T) {
        ep.dstT[(size_t)col * ep.ldt + ep.colbase + row] = (__bf16)v;
      } else if (MODE == EP_XF) {
        v += ep.bias[col]; v = v > 0.f ? v : 0.f;
        float lab = (col < CD) ? ep.h2[(size_t)row * CD + col]
                               : ep.feat[(size_t)row * CD + (col - CD)];
        atomicAdd(ep.outacc + row, v * lab);
      }
    }
  }
}

// ---------------- scalar helper kernels ----------------

__global__ void embed_gather(const int* __restrict__ seq, const float* __restrict__ emb,
                             float* __restrict__ x) {
  int i = blockIdx.x * blockDim.x + threadIdx.x;
  if (i >= CB * CL * CD) return;
  x[i] = emb[(size_t)seq[i / CD] * CD + (i % CD)];
}

__global__ void cvt_pad_bf16(const float* __restrict__ src, int srows, int scols,
                             __bf16* __restrict__ dst, int drows, int dcols) {
  int i = blockIdx.x * blockDim.x + threadIdx.x;
  if (i >= drows * dcols) return;
  int r = i / dcols, c = i % dcols;
  float v = (r < srows && c < scols) ? src[(size_t)r * scols + c] : 0.f;
  dst[i] = (__bf16)v;
}

__global__ void conv_relu(const float* __restrict__ x, const float* __restrict__ Wc,
                          const float* __restrict__ bc, float* __restrict__ c, int ksz) {
  int i = blockIdx.x * blockDim.x + threadIdx.x;   // [b][nk][l], l padded to 512
  if (i >= CB * CNK * CL) return;
  int l  = i & (CL - 1);
  int nk = (i >> 9) & (CNK - 1);
  int b  = i >> 16;
  int Lv = CL - ksz + 1;
  float acc = 0.f;
  if (l < Lv) {
    acc = bc[nk];
    const float* xp = x + ((size_t)(b * CL + l)) * CD;
    const float* wp = Wc + (size_t)nk * ksz * CD;
    for (int j = 0; j < ksz; ++j)
      for (int dd = 0; dd < CD; ++dd)
        acc += xp[j * CD + dd] * wp[j * CD + dd];
    acc = acc > 0.f ? acc : 0.f;
  }
  c[i] = acc;
}

__global__ void xd_tanh(const float* __restrict__ c, const float* __restrict__ Wt,
                        const float* __restrict__ bt, __bf16* __restrict__ xd, int Lv) {
  int i = blockIdx.x * blockDim.x + threadIdx.x;   // [b][l][dp], dp padded to 224
  if (i >= CB * CL * CDP) return;
  int dd = i % CDP;
  int l  = (i / CDP) % CL;
  int b  = i / (CDP * CL);
  float v = 0.f;
  if (dd < CD && l < Lv) {
    float acc = bt[dd];
    const float* cp = c + (size_t)b * CNK * CL + l;  // c[b][k][l], stride CL per k
    const float* wp = Wt + (size_t)dd * CNK;         // Wt[d][k]
    for (int k2 = 0; k2 < CNK; ++k2) acc += cp[(size_t)k2 * CL] * wp[k2];
    v = tanhf(acc);
  }
  xd[i] = (__bf16)v;
}

// column softmax over l (axis=1), writes attention TRANSPOSED in bf16: attT[n][l]
__global__ void softmax_col(const float* __restrict__ S, int Lv,
                            __bf16* __restrict__ attT) {
  int n = blockIdx.x * blockDim.x + threadIdx.x;
  if (n >= CN) return;
  float m = -1e30f;
  for (int l = 0; l < Lv; ++l) { float v = S[(size_t)l * CN + n]; m = v > m ? v : m; }
  float ssum = 0.f;
  for (int l = 0; l < Lv; ++l) ssum += __expf(S[(size_t)l * CN + n] - m);
  float inv = 1.f / ssum;
  for (int l = 0; l < CL; ++l) {
    float a = (l < Lv) ? __expf(S[(size_t)l * CN + n] - m) * inv : 0.f;
    attT[(size_t)n * CL + l] = (__bf16)a;
  }
}

__global__ void scatter_add(const int* __restrict__ s, const int* __restrict__ d,
                            const float* __restrict__ vals, float* __restrict__ out,
                            long long total, int dim) {
  long long i = (long long)blockIdx.x * blockDim.x + threadIdx.x;
  if (i >= total) return;
  int e = (int)(i / dim), c = (int)(i % dim);
  atomicAdd(out + (size_t)d[e] * dim + c, vals[(size_t)s[e] * dim + c]);
}

__global__ void sigmoid_out(const float* __restrict__ acc, float* __restrict__ out, int total) {
  int i = blockIdx.x * blockDim.x + threadIdx.x;
  if (i >= total) return;
  out[i] = 1.f / (1.f + __expf(-acc[i]));
}

// ---------------- host-side launch ----------------
static inline unsigned cdivu(long long a, int b) { return (unsigned)((a + b - 1) / b); }

extern "C" void kernel_launch(void* const* d_in, const int* in_sizes, int n_in,
                              void* d_out, int out_size, void* d_ws, size_t ws_size,
                              hipStream_t stream) {
  const int*   input_seq = (const int*)d_in[0];
  const int*   src       = (const int*)d_in[1];
  const int*   dst       = (const int*)d_in[2];
  const float* features  = (const float*)d_in[3];
  const float* feat      = (const float*)d_in[4];
  const float* emb       = (const float*)d_in[5];
  const float* Wc[3]     = {(const float*)d_in[6], (const float*)d_in[8], (const float*)d_in[10]};
  const float* bc[3]     = {(const float*)d_in[7], (const float*)d_in[9], (const float*)d_in[11]};
  const float* Wt        = (const float*)d_in[12];
  const float* bt        = (const float*)d_in[13];
  const float* Wf        = (const float*)d_in[14];
  const float* bfv       = (const float*)d_in[15];
  const float* Wg1       = (const float*)d_in[16];
  const float* bg1       = (const float*)d_in[17];
  const float* Wg2       = (const float*)d_in[18];
  const float* bg2       = (const float*)d_in[19];
  const int    E         = in_sizes[1];
  const int    KSZ[3]    = {3, 4, 5};

  // ---- workspace carve-up ----
  char* ws = (char*)d_ws;
  size_t off = 0;
  auto take = [&](size_t bytes) -> char* {
    char* p = ws + off; off = (off + bytes + 255) & ~(size_t)255; return p;
  };
  __bf16* feat_bf = (__bf16*)take((size_t)CN * CDP * 2);
  float*  x       = (float*) take((size_t)CB * CL * CD * 4);
  float*  cbuf    = (float*) take((size_t)CB * CNK * CL * 4);
  __bf16* c_bf    = (__bf16*)take((size_t)CB * CNK * CL * 2);
  __bf16* xd_bf   = (__bf16*)take((size_t)CB * CL * CDP * 2);
  float*  S       = (float*) take((size_t)CL * CN * 4);           // per (size,b), reused
  __bf16* attT    = (__bf16*)take((size_t)CN * CL * 2);           // reused
  __bf16* xcT     = (__bf16*)take((size_t)CB * CN * C3NK * 2);
  __bf16* Wf_bf   = (__bf16*)take((size_t)C2D * C3NK * 2);
  __bf16* Wg1_bf  = (__bf16*)take((size_t)CHG * CDP * 2);
  __bf16* Wg2_bf  = (__bf16*)take((size_t)C2DP * CHG * 2);
  float*  agg1    = (float*) take((size_t)CN * CD * 4);
  __bf16* agg1_bf = (__bf16*)take((size_t)CN * CDP * 2);
  float*  h       = (float*) take((size_t)CN * CHG * 4);
  float*  agg2    = (float*) take((size_t)CN * CHG * 4);
  __bf16* agg2_bf = (__bf16*)take((size_t)CN * CHG * 2);
  float*  h2      = (float*) take((size_t)CN * CD * 4);
  float*  outacc  = (float*) take((size_t)CB * CN * 4);

  const int T = 256;
  auto gx = [](int ncols) { return (unsigned)((ncols + 63) / 64); };
  EpArgs ep{};

  // ================= GCN path (produces label = [h2 | feat]) =================
  hipMemsetAsync(agg1, 0, (size_t)CN * CD * 4, stream);
  {
    long long tot = (long long)E * CD;
    scatter_add<<<cdivu(tot, T), T, 0, stream>>>(src, dst, features, agg1, tot, CD);
  }
  cvt_pad_bf16<<<cdivu((long long)CN * CDP, T), T, 0, stream>>>(agg1, CN, CD, agg1_bf, CN, CDP);
  cvt_pad_bf16<<<cdivu((long long)CHG * CDP, T), T, 0, stream>>>(Wg1, CHG, CD, Wg1_bf, CHG, CDP);
  ep = EpArgs{}; ep.dstF = h; ep.ldd = CHG; ep.bias = bg1;
  wmma_gemm_bf16<EP_BIAS_RELU_F32><<<dim3(gx(CHG), CN / 16), 128, 0, stream>>>(
      agg1_bf, CDP, Wg1_bf, CDP, CDP, CHG, ep);

  hipMemsetAsync(agg2, 0, (size_t)CN * CHG * 4, stream);
  {
    long long tot = (long long)E * CHG;
    scatter_add<<<cdivu(tot, T), T, 0, stream>>>(src, dst, h, agg2, tot, CHG);
  }
  cvt_pad_bf16<<<cdivu((long long)CN * CHG, T), T, 0, stream>>>(agg2, CN, CHG, agg2_bf, CN, CHG);
  cvt_pad_bf16<<<cdivu((long long)C2DP * CHG, T), T, 0, stream>>>(Wg2, CD, CHG, Wg2_bf, C2DP, CHG);
  ep = EpArgs{}; ep.dstF = h2; ep.ldd = CD; ep.bias = bg2; ep.guard = CD;
  wmma_gemm_bf16<EP_BIAS_GUARD_F32><<<dim3(gx(C2DP), CN / 16), 128, 0, stream>>>(
      agg2_bf, CHG, Wg2_bf, CHG, CHG, C2DP, ep);

  // ================= CNN-attention content path =================
  embed_gather<<<cdivu((long long)CB * CL * CD, T), T, 0, stream>>>(input_seq, emb, x);
  cvt_pad_bf16<<<cdivu((long long)CN * CDP, T), T, 0, stream>>>(feat, CN, CD, feat_bf, CN, CDP);
  cvt_pad_bf16<<<cdivu((long long)C2D * C3NK, T), T, 0, stream>>>(Wf, C2D, C3NK, Wf_bf, C2D, C3NK);

  for (int si = 0; si < 3; ++si) {
    const int k  = KSZ[si];
    const int Lv = CL - k + 1;
    conv_relu<<<cdivu((long long)CB * CNK * CL, T), T, 0, stream>>>(x, Wc[si], bc[si], cbuf, k);
    cvt_pad_bf16<<<cdivu((long long)CB * CNK * CL, T), T, 0, stream>>>(
        cbuf, CB * CNK, CL, c_bf, CB * CNK, CL);
    xd_tanh<<<cdivu((long long)CB * CL * CDP, T), T, 0, stream>>>(cbuf, Wt, bt, xd_bf, Lv);

    for (int b = 0; b < CB; ++b) {
      // scores: S[l][n] = xd[b,l,:] . feat[n,:]
      ep = EpArgs{}; ep.dstF = S; ep.ldd = CN;
      wmma_gemm_bf16<EP_F32><<<dim3(gx(CN), CL / 16), 128, 0, stream>>>(
          xd_bf + (size_t)b * CL * CDP, CDP, feat_bf, CDP, CDP, CN, ep);
      // softmax over l -> attT[n][l] (bf16)
      softmax_col<<<cdivu(CN, T), T, 0, stream>>>(S, Lv, attT);
      // content: xcT[b][n][si*NK + k'] = sum_l c[b,k',l] * att[l,n]
      ep = EpArgs{}; ep.dstT = xcT + (size_t)b * CN * C3NK; ep.ldt = C3NK; ep.colbase = si * CNK;
      wmma_gemm_bf16<EP_BF16T><<<dim3(gx(CN), CNK / 16), 128, 0, stream>>>(
          c_bf + (size_t)b * CNK * CL, CL, attT, CL, CL, CN, ep);
    }
  }

  // ================= fused x_feature * label dot + sigmoid =================
  hipMemsetAsync(outacc, 0, (size_t)CB * CN * 4, stream);
  for (int b = 0; b < CB; ++b) {
    ep = EpArgs{}; ep.bias = bfv; ep.h2 = h2; ep.feat = feat; ep.outacc = outacc + (size_t)b * CN;
    wmma_gemm_bf16<EP_XF><<<dim3(gx(C2D), CN / 16), 128, 0, stream>>>(
        xcT + (size_t)b * CN * C3NK, C3NK, Wf_bf, C3NK, C3NK, C2D, ep);
  }
  sigmoid_out<<<cdivu((long long)CB * CN, T), T, 0, stream>>>(outacc, (float*)d_out, CB * CN);
}
